// SparseAttention_50208167690716
// MI455X (gfx1250) — matrix operand order repaired and checked by reading, not compiled, not run on hardware
//
#include <hip/hip_runtime.h>
#include <hip/hip_bf16.h>

// ---- problem constants (match the reference) ----
#define NB   4
#define NS   2048
#define ND   768
#define NH   12
#define NHD  64
#define NQB  32          // NS / 64 blocks per sequence

typedef __attribute__((ext_vector_type(16))) _Float16 v16h;
typedef __attribute__((ext_vector_type(2)))  _Float16 h2t;
typedef __attribute__((ext_vector_type(8)))  float    v8f;
typedef __attribute__((ext_vector_type(4)))  unsigned int u32x4;
typedef __attribute__((ext_vector_type(8)))  int      i32x8;
typedef __attribute__((ext_vector_type(4)))  int      i32x4;

union AF { v16h v; h2t p[8]; };

// 16-bit A-matrix 16x32 layout: lane<16 -> K = {2j,2j+1 | j<4} U {8+2j | j>=4};
// lanes 16-31 add +8.  (CDNA5 ISA §7.12.2)
__device__ __forceinline__ int a_koff(int j, int half) {
  return (j < 4 ? 2 * j : 8 + 2 * j) + 8 * half;
}

__device__ __forceinline__ v8f wmma16(v16h a, v16h b, v8f c) {
  return __builtin_amdgcn_wmma_f32_16x16x32_f16(false, a, false, b, (short)0, c,
                                                false, false);
}

// ------------------------------------------------------------------
// TDM: DMA one 64x64 f16 block (contiguous 8192B in global) into LDS,
// padding the destination by 1 dword per 32-dword (128B) row so the LDS
// image has row stride 66 halves (bank-conflict-free column access).
// D# packing per CDNA5 ISA.
// This toolchain exposes the 6-arg builtin (extra i32x8 group, zeroed).
// ------------------------------------------------------------------
__device__ __forceinline__ void tdm_load_block64(const _Float16* gsrc,
                                                 unsigned int lds_off) {
  unsigned long long ga = (unsigned long long)(uintptr_t)gsrc;
  u32x4 g0;
  g0[0] = 1u;                                             // count=1, user D#
  g0[1] = lds_off;                                        // lds_addr
  g0[2] = (unsigned int)ga;                               // global_addr[31:0]
  g0[3] = (unsigned int)((ga >> 32) & 0x1FFFFFFu)         // global_addr[56:32]
        | (2u << 30);                                     // type = 2 (image)
  i32x8 g1;
  g1[0] = (1 << 16)      // data_size = 2 bytes
        | (1 << 20)      // pad_enable
        | (4 << 22);     // pad_interval: 32 dwords (=128B row); pad_amount=0 -> 1 dword
  g1[1] = (64 << 16);    // tensor_dim0 = 64 (low 16 bits at [63:48])
  g1[2] = (64 << 16);    // tensor_dim1 = 64
  g1[3] = (64 << 16);    // tile_dim0  = 64
  g1[4] = 64;            // tile_dim1  = 64, tile_dim2 = 0
  g1[5] = 64;            // tensor_dim0_stride = 64 elements
  g1[6] = 0;
  g1[7] = 0;
  i32x4 z4 = {0, 0, 0, 0};
  i32x8 z8 = {0, 0, 0, 0, 0, 0, 0, 0};
  __builtin_amdgcn_tensor_load_to_lds(g0, g1, z4, z4, z8, 0);
}

// ------------------------------------------------------------------
// Kernel 0: block-sparse index build (numpy bool mask, 1 byte/elem;
// blocks are uniform 64x64, so one probe per block suffices).
// ------------------------------------------------------------------
__global__ void build_blocks_kernel(const unsigned char* __restrict__ mask,
                                    int* __restrict__ cnt, int* __restrict__ list) {
  int qb = threadIdx.x;
  if (qb >= NQB) return;
  int c = 0;
  for (int kb = 0; kb < NQB; ++kb)
    if (mask[(size_t)(qb * 64) * NS + kb * 64]) list[qb * NQB + c++] = kb;
  cnt[qb] = c;
}

// ------------------------------------------------------------------
// Kernel 1: fused QKV projection. Y = X @ W^T, f32 in -> f16 out scattered
// into [B, H, S, 64] head layout. 64x64 tile / WG, K-step 64 (8 WMMA/step).
// grid = (M/64, D/64, 3)  (z selects Wq/Wk/Wv)
// ------------------------------------------------------------------
__global__ __launch_bounds__(128) void proj_qkv_kernel(
    const float* __restrict__ X, const float* __restrict__ Wq,
    const float* __restrict__ Wk, const float* __restrict__ Wv,
    _Float16* __restrict__ Qh, _Float16* __restrict__ Kh,
    _Float16* __restrict__ Vh) {
  __shared__ _Float16 sX[64][66];   // pad 66: column bank-stride 33 (coprime 64)
  __shared__ _Float16 sW[64][66];

  const float* W  = (blockIdx.z == 0) ? Wq : (blockIdx.z == 1) ? Wk : Wv;
  _Float16*  Out  = (blockIdx.z == 0) ? Qh : (blockIdx.z == 1) ? Kh : Vh;
  const int m0 = blockIdx.x * 64;
  const int n0 = blockIdx.y * 64;
  const int tid = threadIdx.x;
  const int lane = tid & 31, wave = tid >> 5;
  const int half = (lane >> 4) & 1, l15 = lane & 15;

  const v8f vzero = {0.f, 0.f, 0.f, 0.f, 0.f, 0.f, 0.f, 0.f};
  v8f acc[4];
#pragma unroll
  for (int t = 0; t < 4; ++t) acc[t] = vzero;

  const int lrow = tid >> 1, lcol = (tid & 1) * 32;
  for (int k0 = 0; k0 < ND; k0 += 64) {
    __syncthreads();
#pragma unroll 8
    for (int i = 0; i < 32; ++i)
      sX[lrow][lcol + i] = (_Float16)X[(size_t)(m0 + lrow) * ND + k0 + lcol + i];
#pragma unroll 8
    for (int i = 0; i < 32; ++i)
      sW[lrow][lcol + i] = (_Float16)W[(size_t)(n0 + lrow) * ND + k0 + lcol + i];
    __syncthreads();

    AF a0, a1;
#pragma unroll
    for (int j = 0; j < 8; ++j) {
      int ka = a_koff(j, half);
      a0.p[j] = *(const h2t*)&sX[wave * 16 + l15][ka];
      a1.p[j] = *(const h2t*)&sX[wave * 16 + l15][32 + ka];
    }
#pragma unroll
    for (int t = 0; t < 4; ++t) {
      AF bw0, bw1;                           // B[k][n] = W[n][k]
#pragma unroll
      for (int j = 0; j < 8; ++j) {
        int kb = a_koff(j, half);
        bw0.p[j] = *(const h2t*)&sW[t * 16 + l15][kb];
        bw1.p[j] = *(const h2t*)&sW[t * 16 + l15][32 + kb];
      }
      acc[t] = wmma16(a0.v, bw0.v, acc[t]);
      acc[t] = wmma16(a1.v, bw1.v, acc[t]);
    }
  }

  // C layout: slot r, lane -> (M = r + 8*half, N = l15)
#pragma unroll
  for (int t = 0; t < 4; ++t)
#pragma unroll
    for (int r = 0; r < 8; ++r) {
      int mg = m0 + wave * 16 + r + 8 * half;     // token index b*S+s
      int ng = n0 + t * 16 + l15;                 // feature = h*64+d
      int bb = mg >> 11, s = mg & (NS - 1);
      int hh = ng >> 6,  d = ng & 63;
      Out[((((size_t)bb * NH + hh) * NS + s) << 6) + d] = (_Float16)acc[t][r];
    }
}

// ------------------------------------------------------------------
// Kernel 2: block-sparse flash attention with TDM double-buffered K/V
// staging. grid = (32 qblocks, H, B); 128 threads = 4 waves, wave owns
// 16 query rows. Wave 0 drives the Tensor Data Mover; DMA for block
// it+1 overlaps WMMA/softmax on block it.
// ------------------------------------------------------------------
__global__ __launch_bounds__(128) void attn_kernel(
    const _Float16* __restrict__ Qh, const _Float16* __restrict__ Kh,
    const _Float16* __restrict__ Vh, const int* __restrict__ cnt,
    const int* __restrict__ list, _Float16* __restrict__ Of) {
  __shared__ _Float16 sK[2][64][66];   // natural [kk][d], TDM-padded rows
  __shared__ _Float16 sV[2][64][66];   // natural [kk][d]
  __shared__ _Float16 sP[4][16][66];   // per-wave C->A relayout scratch

  const int qb = blockIdx.x, h = blockIdx.y, b = blockIdx.z;
  const int tid = threadIdx.x;
  const int lane = tid & 31, wave = tid >> 5;
  const int half = (lane >> 4) & 1, l15 = lane & 15;

  const size_t headBase = (((size_t)b * NH + h) * NS) << 6;   // * 64

  // Q fragments: resident in VGPRs for all key blocks (contraction over d=64)
  AF aq0, aq1;
  {
    const size_t qrow = headBase + (((size_t)qb * 64 + wave * 16 + l15) << 6);
#pragma unroll
    for (int j = 0; j < 8; ++j) {
      int k0 = a_koff(j, half);
      aq0.p[j] = *(const h2t*)&Qh[qrow + k0];
      aq1.p[j] = *(const h2t*)&Qh[qrow + 32 + k0];
    }
  }

  const v8f vzero = {0.f, 0.f, 0.f, 0.f, 0.f, 0.f, 0.f, 0.f};
  v8f oacc[4];
#pragma unroll
  for (int t = 0; t < 4; ++t) oacc[t] = vzero;
  float m_r[8], l_r[8];
#pragma unroll
  for (int r = 0; r < 8; ++r) { m_r[r] = -1e30f; l_r[r] = 0.f; }

  const int nblk = cnt[qb];                   // workgroup-uniform

  // prologue: DMA block 0 into buffer 0
  if (wave == 0 && nblk > 0) {
    int kb0 = list[qb * NQB];
    tdm_load_block64(Kh + headBase + ((size_t)kb0 << 12),
                     (unsigned int)(uintptr_t)&sK[0][0][0]);
    tdm_load_block64(Vh + headBase + ((size_t)kb0 << 12),
                     (unsigned int)(uintptr_t)&sV[0][0][0]);
  }

  int p = 0;
  for (int it = 0; it < nblk; ++it) {
    if (wave == 0) __builtin_amdgcn_s_wait_tensorcnt(0);  // buffer p landed
    __syncthreads();                                      // publish; p^1 readers done
    if (wave == 0 && it + 1 < nblk) {                     // prefetch next block
      int kbn = list[qb * NQB + it + 1];
      tdm_load_block64(Kh + headBase + ((size_t)kbn << 12),
                       (unsigned int)(uintptr_t)&sK[p ^ 1][0][0]);
      tdm_load_block64(Vh + headBase + ((size_t)kbn << 12),
                       (unsigned int)(uintptr_t)&sV[p ^ 1][0][0]);
    }

    // scores S = (Q K^T) / 8 : 16 q-rows x 64 keys per wave (8 WMMAs)
    v8f sc[4];
#pragma unroll
    for (int t = 0; t < 4; ++t) {
      AF bk0, bk1;                            // B[d][kk] = K[kk][d]
#pragma unroll
      for (int j = 0; j < 8; ++j) {
        int kd = a_koff(j, half);
        bk0.p[j] = *(const h2t*)&sK[p][t * 16 + l15][kd];
        bk1.p[j] = *(const h2t*)&sK[p][t * 16 + l15][32 + kd];
      }
      v8f c0 = wmma16(aq0.v, bk0.v, vzero);
      sc[t]  = wmma16(aq1.v, bk1.v, c0);
    }
#pragma unroll
    for (int t = 0; t < 4; ++t)
#pragma unroll
      for (int r = 0; r < 8; ++r) sc[t][r] *= 0.125f;   // 1/sqrt(64)

    // online softmax: row (r + 8*half) lives across lanes of this half-wave
    float lm[8];
#pragma unroll
    for (int r = 0; r < 8; ++r)
      lm[r] = fmaxf(fmaxf(sc[0][r], sc[1][r]), fmaxf(sc[2][r], sc[3][r]));
#pragma unroll
    for (int off = 1; off < 16; off <<= 1)
#pragma unroll
      for (int r = 0; r < 8; ++r)
        lm[r] = fmaxf(lm[r], __shfl_xor(lm[r], off, 32));

    float alpha[8];
#pragma unroll
    for (int r = 0; r < 8; ++r) {
      float mn = fmaxf(m_r[r], lm[r]);
      alpha[r] = __expf(m_r[r] - mn);
      m_r[r] = mn;
    }

    float ls[8];
#pragma unroll
    for (int r = 0; r < 8; ++r) ls[r] = 0.f;
#pragma unroll
    for (int t = 0; t < 4; ++t)
#pragma unroll
      for (int r = 0; r < 8; ++r) {
        float pv = __expf(sc[t][r] - m_r[r]);
        ls[r] += pv;
        sP[wave][r + 8 * half][t * 16 + l15] = (_Float16)pv;  // C-layout store
      }
#pragma unroll
    for (int off = 1; off < 16; off <<= 1)
#pragma unroll
      for (int r = 0; r < 8; ++r)
        ls[r] += __shfl_xor(ls[r], off, 32);
#pragma unroll
    for (int r = 0; r < 8; ++r) l_r[r] = l_r[r] * alpha[r] + ls[r];
#pragma unroll
    for (int t = 0; t < 4; ++t)
#pragma unroll
      for (int r = 0; r < 8; ++r) oacc[t][r] *= alpha[r];

    // reload P in A-layout (same-wave LDS RAW; DS ops complete in order)
    AF ap0, ap1;
#pragma unroll
    for (int j = 0; j < 8; ++j) {
      int k0 = a_koff(j, half);
      ap0.p[j] = *(const h2t*)&sP[wave][l15][k0];
      ap1.p[j] = *(const h2t*)&sP[wave][l15][32 + k0];
    }

    // O += P V  (8 WMMAs). B[kk][d] = V[kk][d]: contraction pairs span two
    // rows of natural sV -> gather as two u16 loads (lanes walk d: no bank
    // conflicts, row stride 33 dwords).
#pragma unroll
    for (int t = 0; t < 4; ++t) {
      const int dcol = t * 16 + l15;
      AF bv0, bv1;
#pragma unroll
      for (int j = 0; j < 8; ++j) {
        int kk = a_koff(j, half);
        h2t v0; v0[0] = sV[p][kk][dcol];      v0[1] = sV[p][kk + 1][dcol];
        h2t v1; v1[0] = sV[p][32 + kk][dcol]; v1[1] = sV[p][33 + kk][dcol];
        bv0.p[j] = v0;
        bv1.p[j] = v1;
      }
      oacc[t] = wmma16(ap0.v, bv0.v, oacc[t]);
      oacc[t] = wmma16(ap1.v, bv1.v, oacc[t]);
    }
    p ^= 1;
  }

  float inv[8];
#pragma unroll
  for (int r = 0; r < 8; ++r) inv[r] = (l_r[r] > 0.f) ? (1.f / l_r[r]) : 0.f;
#pragma unroll
  for (int t = 0; t < 4; ++t)
#pragma unroll
    for (int r = 0; r < 8; ++r) {
      int s = qb * 64 + wave * 16 + r + 8 * half;
      size_t o = ((size_t)b * NS + s) * ND + h * 64 + t * 16 + l15;
      Of[o] = (_Float16)(oacc[t][r] * inv[r]);   // [B,S,H*64] row-major
    }
}

// ------------------------------------------------------------------
// Kernel 3: output projection. Y(f32) = O(f16) @ Wo^T. grid = (128, 12)
// ------------------------------------------------------------------
__global__ __launch_bounds__(128) void proj_out_kernel(
    const _Float16* __restrict__ A, const float* __restrict__ Wo,
    float* __restrict__ Y) {
  __shared__ _Float16 sX[64][66];
  __shared__ _Float16 sW[64][66];
  const int m0 = blockIdx.x * 64;
  const int n0 = blockIdx.y * 64;
  const int tid = threadIdx.x;
  const int lane = tid & 31, wave = tid >> 5;
  const int half = (lane >> 4) & 1, l15 = lane & 15;

  const v8f vzero = {0.f, 0.f, 0.f, 0.f, 0.f, 0.f, 0.f, 0.f};
  v8f acc[4];
#pragma unroll
  for (int t = 0; t < 4; ++t) acc[t] = vzero;

  const int lrow = tid >> 1, lcol = (tid & 1) * 32;
  for (int k0 = 0; k0 < ND; k0 += 64) {
    __syncthreads();
#pragma unroll 8
    for (int i = 0; i < 32; ++i)
      sX[lrow][lcol + i] = A[(size_t)(m0 + lrow) * ND + k0 + lcol + i];
#pragma unroll 8
    for (int i = 0; i < 32; ++i)
      sW[lrow][lcol + i] = (_Float16)Wo[(size_t)(n0 + lrow) * ND + k0 + lcol + i];
    __syncthreads();

    AF a0, a1;
#pragma unroll
    for (int j = 0; j < 8; ++j) {
      int ka = a_koff(j, half);
      a0.p[j] = *(const h2t*)&sX[wave * 16 + l15][ka];
      a1.p[j] = *(const h2t*)&sX[wave * 16 + l15][32 + ka];
    }
#pragma unroll
    for (int t = 0; t < 4; ++t) {
      AF bw0, bw1;
#pragma unroll
      for (int j = 0; j < 8; ++j) {
        int kb = a_koff(j, half);
        bw0.p[j] = *(const h2t*)&sW[t * 16 + l15][kb];
        bw1.p[j] = *(const h2t*)&sW[t * 16 + l15][32 + kb];
      }
      acc[t] = wmma16(a0.v, bw0.v, acc[t]);
      acc[t] = wmma16(a1.v, bw1.v, acc[t]);
    }
  }

#pragma unroll
  for (int t = 0; t < 4; ++t)
#pragma unroll
    for (int r = 0; r < 8; ++r) {
      int mg = m0 + wave * 16 + r + 8 * half;
      int ng = n0 + t * 16 + l15;
      Y[(size_t)mg * ND + ng] = acc[t][r];
    }
}

// ------------------------------------------------------------------
extern "C" void kernel_launch(void* const* d_in, const int* in_sizes, int n_in,
                              void* d_out, int out_size, void* d_ws, size_t ws_size,
                              hipStream_t stream) {
  const float* X  = (const float*)d_in[0];
  const float* Wq = (const float*)d_in[1];
  const float* Wk = (const float*)d_in[2];
  const float* Wv = (const float*)d_in[3];
  const float* Wo = (const float*)d_in[4];
  const unsigned char* mask = (const unsigned char*)d_in[5];  // numpy bool = 1B
  float* Y = (float*)d_out;

  // workspace layout: Qh | Kh | Vh | Of (f16) | cnt | list  (~50.3 MB)
  char* base = (char*)d_ws;
  const size_t elems = (size_t)NB * NS * ND;          // 6,291,456
  _Float16* Qh = (_Float16*)base;
  _Float16* Kh = Qh + elems;
  _Float16* Vh = Kh + elems;
  _Float16* Of = Vh + elems;
  int* cnt  = (int*)(base + 4 * elems * sizeof(_Float16));
  int* list = cnt + NQB;

  build_blocks_kernel<<<1, 32, 0, stream>>>(mask, cnt, list);
  proj_qkv_kernel<<<dim3((NB * NS) / 64, ND / 64, 3), 128, 0, stream>>>(
      X, Wq, Wk, Wv, Qh, Kh, Vh);
  attn_kernel<<<dim3(NQB, NH, NB), 128, 0, stream>>>(Qh, Kh, Vh, cnt, list, Of);
  proj_out_kernel<<<dim3((NB * NS) / 64, ND / 64), 128, 0, stream>>>(Of, Wo, Y);
}
